// loTeNet_75136157876459
// MI455X (gfx1250) — matrix operand order repaired and checked by reading, not compiled, not run on hardware
//
#include <hip/hip_runtime.h>

typedef __attribute__((ext_vector_type(16))) __bf16 v16bf;
typedef __attribute__((ext_vector_type(8)))  float  v8f;

__device__ __forceinline__ __bf16 to_bf(float f) { return (__bf16)f; }

// ---------------------------------------------------------------------------
// Per-patch MPS block + fused BatchNorm.
// One workgroup (128 threads / 4 waves) handles one patch p:
//   m_s[b]  = sum_f phi[b,f,s] * cores[p,s,:,f,:]      (WMMA bf16, K=2C<=32)
//   u0 = row0(m0), u3 = col0(m3), v = u0*m1,
//   out[b,o] = (v * label[p,:,o,:]) * m2 * u3
//   BN over the 512 outputs of this patch (patch == BN channel).
// ---------------------------------------------------------------------------
template <int C, int P, int TRANS>
__global__ void __launch_bounds__(128) mps_block_kernel(
    const float* __restrict__ src, const float* __restrict__ cores,
    const float* __restrict__ label, const float* __restrict__ gamma,
    const float* __restrict__ beta, float* __restrict__ dst)
{
    constexpr int twoC = 2 * C;
    extern __shared__ float sm[];
    const int tid  = threadIdx.x;
    const int lane = tid & 31;
    const int wid  = __builtin_amdgcn_readfirstlane(tid >> 5);   // SGPR: uniform
    const int hi   = lane >> 4;                                  // lane-half
    const int ln   = lane & 15;
    const int p    = blockIdx.x;

    float* xs   = sm;                 // 32*C*4 raw inputs for this patch
    float* m1   = xs + 128 * C;       // 32 x 256  (site-1 matrices, f32)
    // m2 = m1 + 8192                 // 32 x 256  (site-2 matrices)
    float* u0   = m1 + 2 * 8192;      // 32 x 16   row0 of site-0
    float* u3   = u0 + 512;           // 32 x 16   col0 of site-3
    float* vv   = u3 + 512;           // 32 x 16   v = u0*m1
    float* lab  = vv + 512;           // 16x16x16  label core
    float* outv = lab + 4096;         // 32 x 16   patch outputs
    float* red  = outv + 512;         // 128
    float* red2 = red + 128;          // 128

    // ---- stage inputs + label core into LDS ----
    const int cnt = 32 * C * 4;
    for (int i = tid; i < cnt; i += 128) {
        int b = i / (C * 4);
        int r = i - b * (C * 4);                 // r = c*4 + s
        xs[i] = src[(size_t)(b * C + (r >> 2)) * (size_t)(4 * P)
                    + (size_t)p * 4 + (r & 3)];
    }
    for (int i = tid; i < 4096; i += 128)
        lab[i] = label[(size_t)p * 4096 + i];
    __syncthreads();

    // ================= WMMA phase (branchless / wave-uniform) ===============
    // A fragment builder: rows = batch tile mt, K = feature f (zero-pad > 2C)
    auto buildA = [&](int site, int mt) {
        v16bf a;
        int brow = mt * 16 + ln;
#pragma unroll
        for (int j = 0; j < 16; ++j) {
            int f = (j & 7) + (hi << 3) + ((j >> 3) << 4);
            float v;
            if (f < C)          v = xs[(brow * C + f) * 4 + site];
            else if (f < twoC)  v = 1.0f - xs[(brow * C + (f - C)) * 4 + site];
            else                v = 0.0f;
            a[j] = to_bf(v);
        }
        return a;
    };

    // ---- full site-1 / site-2 matrices ----
    // Hoist BOTH batch-tile A fragments per site so each weight (B) fragment is
    // fetched from HBM once and feeds two WMMAs (weights dominate bandwidth).
    for (int site = 1; site <= 2; ++site) {
        const v16bf a0 = buildA(site, 0);
        const v16bf a1 = buildA(site, 1);
        const float* cw = cores + ((size_t)p * 4 + site) * (size_t)(16 * twoC * 16);
        float* mm = m1 + (size_t)(site - 1) * 8192;
        for (int nt = wid; nt < 16; nt += 4) {
            const int col = nt * 16 + ln, dsel = col >> 4, esel = col & 15;
            v16bf bb;
#pragma unroll
            for (int j = 0; j < 16; ++j) {
                int f = (j & 7) + (hi << 3) + ((j >> 3) << 4);
                bb[j] = (f < twoC) ? to_bf(cw[(dsel * twoC + f) * 16 + esel])
                                   : to_bf(0.0f);
            }
            v8f acc0 = {}, acc1 = {};
            acc0 = __builtin_amdgcn_wmma_f32_16x16x32_bf16(
                       false, a0, false, bb, (short)0, acc0, false, false);
            acc1 = __builtin_amdgcn_wmma_f32_16x16x32_bf16(
                       false, a1, false, bb, (short)0, acc1, false, false);
#pragma unroll
            for (int i = 0; i < 8; ++i) {
                mm[(i + (hi << 3)) * 256 + col]      = acc0[i];
                mm[(16 + i + (hi << 3)) * 256 + col] = acc1[i];
            }
        }
    }

    // ---- boundary vectors: each wave does exactly one 16x16 tile ----
    {
        const int isU3 = wid >> 1;                 // waves 0,1->u0; 2,3->u3
        const int mt   = wid & 1;
        const int site = isU3 ? 3 : 0;
        const v16bf a  = buildA(site, mt);
        const float* cw = cores + ((size_t)p * 4 + site) * (size_t)(16 * twoC * 16);
        const int dsel = isU3 ? ln : 0;            // u0: row0 ; u3: col0
        const int esel = isU3 ? 0 : ln;
        v16bf bb;
#pragma unroll
        for (int j = 0; j < 16; ++j) {
            int f = (j & 7) + (hi << 3) + ((j >> 3) << 4);
            bb[j] = (f < twoC) ? to_bf(cw[(dsel * twoC + f) * 16 + esel])
                               : to_bf(0.0f);
        }
        v8f acc = {};
        acc = __builtin_amdgcn_wmma_f32_16x16x32_bf16(
                  false, a, false, bb, (short)0, acc, false, false);
        float* up = u0 + isU3 * 512;
#pragma unroll
        for (int i = 0; i < 8; ++i)
            up[(mt * 16 + i + (hi << 3)) * 16 + ln] = acc[i];
    }
    __syncthreads();

    // ---- v = u0 * m1 (per batch) ----
    float* m2 = m1 + 8192;
    for (int i = tid; i < 512; i += 128) {
        int b = i >> 4, e = i & 15;
        float s = 0.0f;
#pragma unroll
        for (int d = 0; d < 16; ++d) s += u0[b * 16 + d] * m1[b * 256 + d * 16 + e];
        vv[i] = s;
    }
    __syncthreads();

    // ---- chain: out[b,o] = (v * label_o) * m2 * u3 ; BN partials ----
    float ps = 0.0f, ps2 = 0.0f;
    for (int i = tid; i < 512; i += 128) {
        int b = i >> 4, o = i & 15;
        float w[16];
#pragma unroll
        for (int e = 0; e < 16; ++e) {
            float s = 0.0f;
#pragma unroll
            for (int d = 0; d < 16; ++d) s += vv[b * 16 + d] * lab[d * 256 + o * 16 + e];
            w[e] = s;
        }
        float acc = 0.0f;
#pragma unroll
        for (int e = 0; e < 16; ++e) {
            float s = 0.0f;
#pragma unroll
            for (int d = 0; d < 16; ++d) s += w[d] * m2[b * 256 + d * 16 + e];
            acc += s * u3[b * 16 + e];
        }
        outv[i] = acc; ps += acc; ps2 += acc * acc;
    }
    red[tid] = ps; red2[tid] = ps2;
    __syncthreads();
    for (int off = 64; off > 0; off >>= 1) {
        if (tid < off) { red[tid] += red[tid + off]; red2[tid] += red2[tid + off]; }
        __syncthreads();
    }
    const float mu  = red[0] * (1.0f / 512.0f);
    const float var = red2[0] * (1.0f / 512.0f) - mu * mu;
    const float rs  = rsqrtf(var + 1e-5f);
    const float g = gamma[p], bt = beta[p];
    for (int i = tid; i < 512; i += 128) {
        int b = i >> 4, o = i & 15;
        float val = g * (outv[i] - mu) * rs + bt;
        if (TRANS) dst[(size_t)b * 4096 + o * 256 + p] = val;           // (b,16,256)
        else       dst[(size_t)b * (size_t)(P * 16) + p * 16 + o] = val;
    }
}

// ---------------------------------------------------------------------------
// Final MPS: build per-site matrices m[n,b] = phi[b,:,n] * coresF[n]  (WMMA).
// ---------------------------------------------------------------------------
__global__ void __launch_bounds__(128) mps_final_build(
    const float* __restrict__ y3, const float* __restrict__ coresF,
    float* __restrict__ mF)
{
    const int tid  = threadIdx.x, lane = tid & 31;
    const int wid  = __builtin_amdgcn_readfirstlane(tid >> 5);
    const int hi   = lane >> 4, ln = lane & 15;
    const int n    = blockIdx.x;

    v16bf a01[2];
#pragma unroll
    for (int mt = 0; mt < 2; ++mt) {
        const int b = mt * 16 + ln;
#pragma unroll
        for (int j = 0; j < 16; ++j) {
            int f = (j & 7) + (hi << 3) + ((j >> 3) << 4);
            float v = (f < 16) ? y3[(size_t)b * 4096 + f * 256 + n]
                               : 1.0f - y3[(size_t)b * 4096 + (f - 16) * 256 + n];
            a01[mt][j] = to_bf(v);
        }
    }
    for (int nt = wid; nt < 16; nt += 4) {
        const int col = nt * 16 + ln, d = col >> 4, e = col & 15;
        v16bf bb;
#pragma unroll
        for (int j = 0; j < 16; ++j) {
            int f = (j & 7) + (hi << 3) + ((j >> 3) << 4);
            bb[j] = to_bf(coresF[(((size_t)n * 16 + d) * 32 + f) * 16 + e]);
        }
        v8f acc0 = {}, acc1 = {};
        acc0 = __builtin_amdgcn_wmma_f32_16x16x32_bf16(
                   false, a01[0], false, bb, (short)0, acc0, false, false);
        acc1 = __builtin_amdgcn_wmma_f32_16x16x32_bf16(
                   false, a01[1], false, bb, (short)0, acc1, false, false);
#pragma unroll
        for (int i = 0; i < 8; ++i) {
            mF[(size_t)n * 8192 + (i + (hi << 3)) * 256 + col]      = acc0[i];
            mF[(size_t)n * 8192 + (16 + i + (hi << 3)) * 256 + col] = acc1[i];
        }
    }
}

// Ordered product of 8 consecutive site matrices (per b, per half-chain).
__global__ void __launch_bounds__(256) chain8(const float* __restrict__ mF,
                                              float* __restrict__ P1)
{
    __shared__ float A[256];
    const int bid = blockIdx.x;              // (b*2 + half)*16 + chunk
    const int b = bid >> 5, half = (bid >> 4) & 1, ck = bid & 15;
    const int n0 = half * 128 + ck * 8;
    const int tid = threadIdx.x, d = tid >> 4, e = tid & 15;
    A[tid] = mF[(size_t)n0 * 8192 + b * 256 + tid];
    __syncthreads();
    for (int i = 1; i < 8; ++i) {
        const float* M = mF + (size_t)(n0 + i) * 8192 + b * 256;
        float v = 0.0f;
#pragma unroll
        for (int k = 0; k < 16; ++k) v += A[d * 16 + k] * M[k * 16 + e];
        __syncthreads();
        A[tid] = v;
        __syncthreads();
    }
    P1[(size_t)bid * 256 + tid] = A[tid];
}

// Ordered product of 16 chunk-products -> L (half 0) / R (half 1) per batch.
__global__ void __launch_bounds__(256) chain16(const float* __restrict__ P1,
                                               float* __restrict__ LR)
{
    __shared__ float A[256];
    const int bid = blockIdx.x;              // b*2 + half
    const int tid = threadIdx.x, d = tid >> 4, e = tid & 15;
    A[tid] = P1[(size_t)bid * 4096 + tid];
    __syncthreads();
    for (int j = 1; j < 16; ++j) {
        const float* M = P1 + (size_t)(bid * 16 + j) * 256;
        float v = 0.0f;
#pragma unroll
        for (int k = 0; k < 16; ++k) v += A[d * 16 + k] * M[k * 16 + e];
        __syncthreads();
        A[tid] = v;
        __syncthreads();
    }
    LR[(size_t)bid * 256 + tid] = A[tid];
}

// out[b,o] = sum_{d,e} L_b[0,d] * labelF[d,o,e] * R_b[e,0]
__global__ void final_out(const float* __restrict__ LR,
                          const float* __restrict__ labelF,
                          float* __restrict__ out)
{
    int t = threadIdx.x;
    if (t < 64) {
        int b = t >> 1, o = t & 1;
        const float* L = LR + (size_t)(b * 2) * 256;
        const float* R = LR + (size_t)(b * 2 + 1) * 256;
        float acc = 0.0f;
#pragma unroll
        for (int d = 0; d < 16; ++d) {
            float ld = L[d];                  // row 0 of L
#pragma unroll
            for (int e = 0; e < 16; ++e)
                acc += ld * labelF[d * 32 + o * 16 + e] * R[e * 16];
        }
        out[b * 2 + o] = acc;
    }
}

extern "C" void kernel_launch(void* const* d_in, const int* in_sizes, int n_in,
                              void* d_out, int out_size, void* d_ws, size_t ws_size,
                              hipStream_t stream)
{
    (void)in_sizes; (void)n_in; (void)out_size; (void)ws_size;
    const float* x      = (const float*)d_in[0];
    const float* cores1 = (const float*)d_in[1];
    const float* label1 = (const float*)d_in[2];
    const float* g1     = (const float*)d_in[3];
    const float* b1     = (const float*)d_in[4];
    const float* cores2 = (const float*)d_in[5];
    const float* label2 = (const float*)d_in[6];
    const float* g2     = (const float*)d_in[7];
    const float* b2     = (const float*)d_in[8];
    const float* cores3 = (const float*)d_in[9];
    const float* label3 = (const float*)d_in[10];
    const float* g3     = (const float*)d_in[11];
    const float* b3     = (const float*)d_in[12];
    const float* coresF = (const float*)d_in[13];
    const float* labelF = (const float*)d_in[14];

    float* ws = (float*)d_ws;
    float* y1 = ws;                         // 32 * 65536
    float* y2 = y1 + (size_t)32 * 65536;    // 32 * 16384
    float* y3 = y2 + (size_t)32 * 16384;    // 32 * 4096  (b,16,256 layout)
    float* mF = y3 + (size_t)32 * 4096;     // 256 * 32 * 256
    float* P1 = mF + (size_t)256 * 8192;    // 64 * 16 * 256
    float* LR = P1 + (size_t)64 * 4096;     // 64 * 256

    const size_t shBase = 22784;            // floats, excluding xs
    size_t sh1 = (shBase + 128 * 3)  * sizeof(float);
    size_t sh2 = (shBase + 128 * 16) * sizeof(float);

    mps_block_kernel<3, 4096, 0><<<4096, 128, sh1, stream>>>(x,  cores1, label1, g1, b1, y1);
    mps_block_kernel<16, 1024, 0><<<1024, 128, sh2, stream>>>(y1, cores2, label2, g2, b2, y2);
    mps_block_kernel<16, 256, 1><<<256,  128, sh2, stream>>>(y2, cores3, label3, g3, b3, y3);
    mps_final_build<<<256, 128, 0, stream>>>(y3, coresF, mF);
    chain8 <<<1024, 256, 0, stream>>>(mF, P1);
    chain16<<<64,   256, 0, stream>>>(P1, LR);
    final_out<<<1, 64, 0, stream>>>(LR, labelF, (float*)d_out);
}
